// AttentionBlock_57380763074683
// MI455X (gfx1250) — matrix operand order repaired and checked by reading, not compiled, not run on hardware
//
#include <hip/hip_runtime.h>
#include <math.h>

#define BATCH 16
#define CCH   256
#define SEQ   1024
#define NH    4
#define DKK   64
#define QKSCALE 0.125f   // 1/sqrt(64)

typedef __attribute__((ext_vector_type(16))) __bf16 v16bf;
typedef __attribute__((ext_vector_type(8)))  __bf16 v8bf;
typedef __attribute__((ext_vector_type(8)))  float  v8f;

// ---------- bf16 helpers (round-to-nearest-even) ----------
__device__ __forceinline__ unsigned short f2bfu(float f) {
  unsigned u = __float_as_uint(f);
  u += 0x7FFFu + ((u >> 16) & 1u);
  return (unsigned short)(u >> 16);
}
__device__ __forceinline__ __bf16 f2bfh(float f) {
  unsigned short h = f2bfu(f);
  return __builtin_bit_cast(__bf16, h);
}

// ---------- WMMA wrapper ----------
__device__ __forceinline__ v8f wmma_bf16(v16bf a, v16bf b, v8f c) {
  return __builtin_amdgcn_wmma_f32_16x16x32_bf16(
      /*neg_a=*/false, a, /*neg_b=*/false, b,
      /*c_mod=*/(short)0, c, /*reuse_a=*/false, /*reuse_b=*/false);
}

// A fragment: A row-major [16][K], tile columns [kbase, kbase+32)
// lane L: m=L&15, kh=L>>4; halves 0-7 = K kbase+kh*8.., halves 8-15 = K kbase+16+kh*8..
__device__ __forceinline__ v16bf load_frag_a(const unsigned short* Au, int ld, int kbase, int lane) {
  const __bf16* A = (const __bf16*)Au;
  int m = lane & 15, kh = lane >> 4;
  const __bf16* p = A + (size_t)m * ld + kbase + kh * 8;
  v8bf lo = *(const v8bf*)p;
  v8bf hi = *(const v8bf*)(p + 16);
  return __builtin_shufflevector(lo, hi, 0,1,2,3,4,5,6,7,8,9,10,11,12,13,14,15);
}

// B fragment: Bt row-major [16 n][K] (B transposed), tile K rows [kbase, kbase+32)
// lane L: n=L&15, kh=L>>4; halves 0-7 = K kbase+kh*8.., halves 8-15 = K kbase+16+kh*8..
__device__ __forceinline__ v16bf load_frag_b(const unsigned short* Btu, int ld, int kbase, int lane) {
  const __bf16* Bt = (const __bf16*)Btu;
  int n = lane & 15, kh = lane >> 4;
  const __bf16* p = Bt + (size_t)n * ld + kbase + kh * 8;
  v8bf lo = *(const v8bf*)p;
  v8bf hi = *(const v8bf*)(p + 16);
  return __builtin_shufflevector(lo, hi, 0,1,2,3,4,5,6,7,8,9,10,11,12,13,14,15);
}

// ---------- prep kernels: convert/transpose to bf16 ----------
__global__ void k_prep_wp(const float* __restrict__ wp, unsigned short* __restrict__ wpT) {
  int i = blockIdx.x * blockDim.x + threadIdx.x;
  if (i >= 768 * 256) return;
  int o = i >> 8, c = i & 255;                 // wpT[o][c] = w_proj[c][o]
  wpT[i] = f2bfu(wp[(size_t)c * 768 + o]);
}
__global__ void k_prep_wo(const float* __restrict__ wo, unsigned short* __restrict__ woT) {
  int i = blockIdx.x * blockDim.x + threadIdx.x;
  if (i >= 256 * 256) return;
  int n = i >> 8, k = i & 255;                 // woT[n][k] = w_out[k][n]
  woT[i] = f2bfu(wo[(size_t)k * 256 + n]);
}
__global__ void k_prep_x(const float* __restrict__ x, unsigned short* __restrict__ xfb) {
  int i = blockIdx.x * blockDim.x + threadIdx.x;
  if (i >= BATCH * SEQ * CCH) return;
  int c = i & 255, s = (i >> 8) & 1023, b = i >> 18;   // xfb[b][s][c] = x[b][c][s]
  xfb[i] = f2bfu(x[((size_t)b * CCH + c) * SEQ + s]);
}

// ---------- QKV projection: Q,K part. grid(S/16, B), block 128 (wave = head) ----------
__global__ __launch_bounds__(128) void k_proj_qk(const unsigned short* __restrict__ xfb,
                                                 const unsigned short* __restrict__ wpT,
                                                 const float* __restrict__ bp,
                                                 unsigned short* __restrict__ Q,
                                                 unsigned short* __restrict__ K) {
  int s0 = blockIdx.x * 16;
  int b  = blockIdx.y;
  int wave = threadIdx.x >> 5;   // head
  int lane = threadIdx.x & 31;
  const unsigned short* A = xfb + (size_t)(b * SEQ + s0) * CCH;
  v8f acc[8];
  #pragma unroll
  for (int nt = 0; nt < 8; ++nt)
    #pragma unroll
    for (int r = 0; r < 8; ++r) acc[nt][r] = 0.f;

  for (int kc = 0; kc < CCH; kc += 32) {
    v16bf a = load_frag_a(A, CCH, kc, lane);
    #pragma unroll
    for (int nt = 0; nt < 8; ++nt) {
      const unsigned short* Bt = wpT + (size_t)(wave * 192 + nt * 16) * CCH;
      v16bf bb = load_frag_b(Bt, CCH, kc, lane);
      acc[nt] = wmma_bf16(a, bb, acc[nt]);
    }
  }
  int hi = lane >> 4, n = lane & 15;
  unsigned short* Qd = Q + (size_t)((b * NH + wave) * SEQ + s0) * DKK;
  unsigned short* Kd = K + (size_t)((b * NH + wave) * SEQ + s0) * DKK;
  #pragma unroll
  for (int nt = 0; nt < 8; ++nt) {
    float bias = bp[wave * 192 + nt * 16 + n];
    unsigned short* dst = (nt < 4) ? Qd : Kd;
    int d = (nt & 3) * 16 + n;
    #pragma unroll
    for (int r = 0; r < 8; ++r) {
      int M = r + hi * 8;
      dst[(size_t)M * DKK + d] = f2bfu(acc[nt][r] + bias);
    }
  }
}

// ---------- QKV projection: V part, transposed out. grid(S/64, H, B), block 128 ----------
__global__ __launch_bounds__(128) void k_proj_v(const unsigned short* __restrict__ xfb,
                                                const unsigned short* __restrict__ wpT,
                                                const float* __restrict__ bp,
                                                unsigned short* __restrict__ Vt) {
  int s0 = blockIdx.x * 64;
  int h  = blockIdx.y;
  int b  = blockIdx.z;
  int wave = threadIdx.x >> 5;   // d-row tile
  int lane = threadIdx.x & 31;
  int orow = h * 192 + 128 + wave * 16;
  const unsigned short* A = wpT + (size_t)orow * CCH;   // A[d][c] = w_proj[c][o]
  v8f acc[4];
  #pragma unroll
  for (int nt = 0; nt < 4; ++nt)
    #pragma unroll
    for (int r = 0; r < 8; ++r) acc[nt][r] = 0.f;

  for (int kc = 0; kc < CCH; kc += 32) {
    v16bf a = load_frag_a(A, CCH, kc, lane);
    #pragma unroll
    for (int nt = 0; nt < 4; ++nt) {
      const unsigned short* Bt = xfb + (size_t)(b * SEQ + s0 + nt * 16) * CCH;
      v16bf bb = load_frag_b(Bt, CCH, kc, lane);
      acc[nt] = wmma_bf16(a, bb, acc[nt]);
    }
  }
  int hi = lane >> 4, n = lane & 15;
  unsigned short* Vd = Vt + (size_t)((b * NH + h) * DKK + wave * 16) * SEQ;
  #pragma unroll
  for (int r = 0; r < 8; ++r) {
    int M = r + hi * 8;
    float bias = bp[orow + M];
    #pragma unroll
    for (int nt = 0; nt < 4; ++nt)
      Vd[(size_t)M * SEQ + s0 + nt * 16 + n] = f2bfu(acc[nt][r] + bias);
  }
}

// P fragment built from fp32 LDS scores (already exp'd), scaled by 1/rowsum
__device__ __forceinline__ v16bf load_p_frag(const float* sc, const float* rinv, int jc, int lane) {
  int m = lane & 15, kh = lane >> 4;
  const float* p = sc + (size_t)m * SEQ + jc + kh * 8;
  float rv = rinv[m];
  v16bf f;
  #pragma unroll
  for (int t = 0; t < 8; ++t) f[t] = f2bfh(p[t] * rv);
  #pragma unroll
  for (int t = 0; t < 8; ++t) f[t + 8] = f2bfh(p[16 + t] * rv);
  return f;
}

// ---------- attention core. grid(S/16, H, B), block 128, dyn LDS 64KB+64B ----------
__global__ __launch_bounds__(128) void k_attn(const unsigned short* __restrict__ Q,
                                              const unsigned short* __restrict__ K,
                                              const unsigned short* __restrict__ Vt,
                                              unsigned short* __restrict__ res) {
  int i0 = blockIdx.x * 16, h = blockIdx.y, b = blockIdx.z;
  int tid = threadIdx.x, wave = tid >> 5, lane = tid & 31;
  extern __shared__ char smem[];
  float* sc   = (float*)smem;                       // [16][SEQ]
  float* rinv = (float*)(smem + 16 * SEQ * 4);      // [16]
  const unsigned short* Qp = Q  + (size_t)((b * NH + h) * SEQ + i0) * DKK;
  const unsigned short* Kp = K  + (size_t)((b * NH + h) * SEQ) * DKK;
  const unsigned short* Vp = Vt + (size_t)((b * NH + h) * DKK) * SEQ;
  int hi = lane >> 4, n = lane & 15;

  // Phase 1: scores = Q K^T * scale -> LDS (wave-strided over j-tiles)
  v16bf aq0 = load_frag_a(Qp, DKK, 0, lane);
  v16bf aq1 = load_frag_a(Qp, DKK, 32, lane);
  for (int jt = wave; jt < SEQ / 16; jt += 4) {
    v8f acc;
    #pragma unroll
    for (int r = 0; r < 8; ++r) acc[r] = 0.f;
    const unsigned short* Bt = Kp + (size_t)jt * 16 * DKK;   // Bt[j][d] = K[j][d]
    acc = wmma_bf16(aq0, load_frag_b(Bt, DKK, 0, lane), acc);
    acc = wmma_bf16(aq1, load_frag_b(Bt, DKK, 32, lane), acc);
    #pragma unroll
    for (int r = 0; r < 8; ++r)
      sc[(size_t)(r + hi * 8) * SEQ + jt * 16 + n] = acc[r] * QKSCALE;
  }
  __syncthreads();

  // Phase 2: softmax rows; wave w owns rows 4w..4w+3, 8 lanes per row
  {
    int row = wave * 4 + (lane >> 3);
    int g = lane & 7;
    float* p = sc + (size_t)row * SEQ;
    float m = -3.4e38f;
    for (int c = g; c < SEQ; c += 8) m = fmaxf(m, p[c]);
    m = fmaxf(m, __shfl_xor(m, 1));
    m = fmaxf(m, __shfl_xor(m, 2));
    m = fmaxf(m, __shfl_xor(m, 4));
    float s = 0.f;
    for (int c = g; c < SEQ; c += 8) { float e = __expf(p[c] - m); p[c] = e; s += e; }
    s += __shfl_xor(s, 1);
    s += __shfl_xor(s, 2);
    s += __shfl_xor(s, 4);
    if (g == 0) rinv[row] = 1.f / s;
  }
  __syncthreads();

  // Phase 3: out = P V ; wave w -> d-tile w. Bt[d][j] = Vt[d][j]
  v8f acc;
  #pragma unroll
  for (int r = 0; r < 8; ++r) acc[r] = 0.f;
  const unsigned short* Bt = Vp + (size_t)(wave * 16) * SEQ;
  for (int jc = 0; jc < SEQ; jc += 32) {
    v16bf ap = load_p_frag(sc, rinv, jc, lane);
    acc = wmma_bf16(ap, load_frag_b(Bt, SEQ, jc, lane), acc);
  }
  unsigned short* rd = res + (size_t)(b * SEQ + i0) * (NH * DKK) + h * DKK + wave * 16;
  #pragma unroll
  for (int r = 0; r < 8; ++r)
    rd[(size_t)(r + hi * 8) * (NH * DKK) + n] = f2bfu(acc[r]);
}

// ---------- out projection + bias + residual, transposed store. grid(S/64, C/16, B), block 32 ----------
__global__ __launch_bounds__(32) void k_outproj(const unsigned short* __restrict__ res,
                                                const unsigned short* __restrict__ woT,
                                                const float* __restrict__ bo,
                                                const float* __restrict__ x,
                                                float* __restrict__ out) {
  int s0 = blockIdx.x * 64, c0 = blockIdx.y * 16, b = blockIdx.z;
  int lane = threadIdx.x & 31;
  const unsigned short* A = woT + (size_t)c0 * CCH;   // A[c][k] = w_out[k][c]
  v8f acc[4];
  #pragma unroll
  for (int nt = 0; nt < 4; ++nt)
    #pragma unroll
    for (int r = 0; r < 8; ++r) acc[nt][r] = 0.f;

  for (int kc = 0; kc < CCH; kc += 32) {
    v16bf a = load_frag_a(A, CCH, kc, lane);
    #pragma unroll
    for (int nt = 0; nt < 4; ++nt) {
      const unsigned short* Bt = res + (size_t)(b * SEQ + s0 + nt * 16) * CCH;
      v16bf bb = load_frag_b(Bt, CCH, kc, lane);
      acc[nt] = wmma_bf16(a, bb, acc[nt]);
    }
  }
  int hi = lane >> 4, n = lane & 15;
  #pragma unroll
  for (int r = 0; r < 8; ++r) {
    int c = c0 + r + hi * 8;
    float bias = bo[c];
    const float* xr = x   + ((size_t)(b * CCH + c)) * SEQ;
    float*       od = out + ((size_t)(b * CCH + c)) * SEQ;
    #pragma unroll
    for (int nt = 0; nt < 4; ++nt) {
      int s = s0 + nt * 16 + n;
      od[s] = acc[nt][r] + bias + xr[s];   // + residual xf
    }
  }
}

extern "C" void kernel_launch(void* const* d_in, const int* in_sizes, int n_in,
                              void* d_out, int out_size, void* d_ws, size_t ws_size,
                              hipStream_t stream) {
  (void)in_sizes; (void)n_in; (void)out_size; (void)ws_size;
  const float* x  = (const float*)d_in[0];
  const float* wp = (const float*)d_in[1];
  const float* bp = (const float*)d_in[2];
  const float* wo = (const float*)d_in[3];
  const float* bo = (const float*)d_in[4];
  float* out = (float*)d_out;

  char* ws = (char*)d_ws;
  size_t off = 0;
  auto take = [&](size_t bytes) -> char* {
    char* p = ws + off;
    off += (bytes + 255) & ~(size_t)255;
    return p;
  };
  unsigned short* xfb = (unsigned short*)take((size_t)BATCH * SEQ * CCH * 2);  // [b][s][c]
  unsigned short* wpT = (unsigned short*)take((size_t)768 * 256 * 2);          // [o][c]
  unsigned short* woT = (unsigned short*)take((size_t)256 * 256 * 2);          // [n][k]
  unsigned short* Q   = (unsigned short*)take((size_t)BATCH * NH * SEQ * DKK * 2);
  unsigned short* K   = (unsigned short*)take((size_t)BATCH * NH * SEQ * DKK * 2);
  unsigned short* Vt  = (unsigned short*)take((size_t)BATCH * NH * DKK * SEQ * 2);
  unsigned short* res = (unsigned short*)take((size_t)BATCH * SEQ * (NH * DKK) * 2);

  k_prep_wp<<<(768 * 256 + 255) / 256, 256, 0, stream>>>(wp, wpT);
  k_prep_wo<<<(256 * 256 + 255) / 256, 256, 0, stream>>>(wo, woT);
  k_prep_x<<<(BATCH * SEQ * CCH + 255) / 256, 256, 0, stream>>>(x, xfb);

  { dim3 g(SEQ / 16, BATCH, 1);  k_proj_qk<<<g, 128, 0, stream>>>(xfb, wpT, bp, Q, K); }
  { dim3 g(SEQ / 64, NH, BATCH); k_proj_v<<<g, 128, 0, stream>>>(xfb, wpT, bp, Vt); }
  { dim3 g(SEQ / 16, NH, BATCH);
    size_t smem = (size_t)16 * SEQ * 4 + 64;
    k_attn<<<g, 128, smem, stream>>>(Q, K, Vt, res); }
  { dim3 g(SEQ / 64, CCH / 16, BATCH); k_outproj<<<g, 32, 0, stream>>>(res, woT, bo, x, out); }
}
